// Annotator_12773232738702
// MI455X (gfx1250) — matrix operand order repaired and checked by reading, not compiled, not run on hardware
//
#include <hip/hip_runtime.h>
#include <hip/hip_bf16.h>

// ---------------- problem constants (from reference) ----------------
#define B_    256
#define S_    16
#define W_    32
#define T_    (S_ * W_)      // 512
#define E_    256
#define H_    512
#define C_    5
#define D1_   2
#define D2_   3
#define KTOT  (E_ + H_)      // 768 : fused K (x | h)
#define G4H   (4 * H_)       // 2048
#define NCHUNK (KTOT / 32)   // 24 K-chunks of 32
#define NCTILE (G4H / 16)    // 128 column tiles of 16

typedef __bf16 v16bf __attribute__((ext_vector_type(16)));
typedef float  v8f   __attribute__((ext_vector_type(8)));

// 16-bit A-matrix 16x32 lane layout (ISA 7.12.2): per lane the fragment is two
// contiguous 16-byte K-runs at K = lo and K = lo+16 (lo = 0 for lanes 0-15, 8 else).
__device__ __forceinline__ int b_k_of(int j, int lane) {
    return ((lane & 16) ? 8 : 0) + (j < 8 ? j : j + 8);   // B: lanes 0-15 K=0..7,16..23; 16-31 K=8..15,24..31
}

__device__ __forceinline__ float sigf(float x) { return 1.0f / (1.0f + expf(-x)); }

union AFrag { v16bf v; uint4 q[2]; };

// ---------------- prep: weights packed in exact WMMA B-fragment order ----------------
__global__ void prep_pack(const float* __restrict__ Wih,
                          const float* __restrict__ Whh,
                          __bf16* __restrict__ Wpack) {
    int tid = blockIdx.x * blockDim.x + threadIdx.x;   // NCHUNK*NCTILE*32 = 98304
    if (tid >= NCHUNK * NCTILE * 32) return;
    int lane  = tid & 31;
    int ctile = (tid >> 5) & (NCTILE - 1);
    int c     = tid >> 12;                 // chunk 0..23
    int kc    = c * 32;
    int col   = ctile * 16 + (lane & 15);

    __bf16 frag[16];
#pragma unroll
    for (int j = 0; j < 16; ++j) {
        int k = kc + b_k_of(j, lane);
        float v = (k < E_) ? Wih[(size_t)col * E_ + k]
                           : Whh[(size_t)col * H_ + (k - E_)];
        frag[j] = (__bf16)v;
    }
    __bf16* dst = Wpack + (size_t)tid * 16;   // 32B contiguous per lane
#pragma unroll
    for (int q = 0; q < 2; ++q)
        *(uint4*)(dst + q * 8) = *(uint4*)(frag + q * 8);
}

// ---------------- prep: one-shot embedding gather + f32->bf16  xbf[T,B,E] ----------------
__global__ void prep_x(const int*   __restrict__ abstracts,  // [B,S,W] = [B,T]
                       const float* __restrict__ emb,        // [V,E]
                       __bf16* __restrict__ xbf) {           // [T,B,E]
    int tid = blockIdx.x * blockDim.x + threadIdx.x;         // T*B*(E/8) = 4,194,304
    if (tid >= T_ * B_ * (E_ / 8)) return;
    int rowid = tid >> 5;                 // (t,b) row ; E/8 = 32 threads per row
    int t  = rowid >> 8;                  // / B_
    int b  = rowid & (B_ - 1);
    int eo = (tid & 31) * 8;
    int idx = abstracts[(size_t)b * T_ + t];

    const float* src = emb + (size_t)idx * E_ + eo;
    float f[8];
    *(float4*)(f + 0) = *(const float4*)(src + 0);
    *(float4*)(f + 4) = *(const float4*)(src + 4);
    __bf16 tmp[8];
#pragma unroll
    for (int j = 0; j < 8; ++j) tmp[j] = (__bf16)f[j];
    *(uint4*)&xbf[((size_t)t * B_ + b) * E_ + eo] = *(uint4*)tmp;
}

// ---------------- init: state, bias, running max ----------------
__global__ void init_state(const float* __restrict__ b_ih,
                           const float* __restrict__ b_hh,
                           float* __restrict__ biasC,
                           __bf16* __restrict__ hbf0,
                           float* __restrict__ cst,
                           float* __restrict__ reps) {
    int tid = blockIdx.x * blockDim.x + threadIdx.x;   // grid covers S*B*H
    if (tid < G4H) biasC[tid] = b_ih[tid] + b_hh[tid];
    if (tid < B_ * H_) { cst[tid] = 0.0f; hbf0[tid] = (__bf16)0.0f; }
    if (tid < S_ * B_ * H_) reps[tid] = -3.402823466e38f;
}

// ---- shared epilogue: gate tiles in LDS -> LSTM cell update + fused max-pool ----
__device__ __forceinline__ void lstm_pointwise(
        const float (*gate_lds)[16 * 32], int b0, int hn, int s,
        const float* __restrict__ biasC, float* __restrict__ cst,
        __bf16* __restrict__ hnext, float* __restrict__ reps) {
#pragma unroll
    for (int i = 0; i < 4; ++i) {
        int e = threadIdx.x + i * 128;           // [0,512)
        int mr = e >> 5, n = e & 31;
        int b = b0 + mr;
        int h = hn * 32 + n;
        float gi = gate_lds[0][e] + biasC[0 * H_ + h];
        float gf = gate_lds[1][e] + biasC[1 * H_ + h];
        float gg = gate_lds[2][e] + biasC[2 * H_ + h];
        float go = gate_lds[3][e] + biasC[3 * H_ + h];
        float cO = cst[(size_t)b * H_ + h];
        float c2 = sigf(gf) * cO + sigf(gi) * tanhf(gg);
        float h2 = sigf(go) * tanhf(c2);
        cst[(size_t)b * H_ + h] = c2;
        hnext[(size_t)b * H_ + h] = (__bf16)h2;
        float* rp = &reps[(size_t)s * B_ * H_ + (size_t)b * H_ + h];
        *rp = fmaxf(*rp, h2);                    // exclusive owner; stream-ordered
    }
}

// ---------------- one LSTM timestep (fast path: pre-converted bf16 x) ----------------
// grid = (B/16)*(H/32) = 256 blocks, 128 threads (4 waves).
// Wave g owns gate g of a 16(b) x 32(h) tile; A fragments built per-lane from
// global bf16 (two b128 loads) -- no LDS, no barriers until the pointwise join.
__global__ void __launch_bounds__(128)
lstm_step_fast(int t,
               const __bf16* __restrict__ xbf,    // [T,B,E] bf16
               const __bf16* __restrict__ Wpack,
               const float*  __restrict__ biasC,
               const __bf16* __restrict__ hbf,
               __bf16* __restrict__ hbf_out_base,
               float* __restrict__ cst,
               float* __restrict__ reps) {
    __shared__ float gate_lds[4][16 * 32];

    const int bm   = blockIdx.x & 15;
    const int hn   = blockIdx.x >> 4;
    const int g    = threadIdx.x >> 5;
    const int lane = threadIdx.x & 31;
    const int b0   = bm * 16;
    const int s    = t / W_;

    const __bf16* hprev = hbf + (size_t)(t & 1) * (B_ * H_);
    __bf16* hnext = hbf_out_base + (size_t)((t + 1) & 1) * (B_ * H_);

    const int m  = lane & 15;
    const int lo = (lane & 16) ? 8 : 0;
    const __bf16* xrow = xbf + ((size_t)t * B_ + b0 + m) * E_ + lo;
    const __bf16* hrow = hprev + (size_t)(b0 + m) * H_ + lo;

    const int ctile0 = g * 32 + hn * 2;
    const __bf16* wb = Wpack + (size_t)ctile0 * (32 * 16) + (size_t)lane * 16;

    v8f acc0 = {0.f, 0.f, 0.f, 0.f, 0.f, 0.f, 0.f, 0.f};
    v8f acc1 = {0.f, 0.f, 0.f, 0.f, 0.f, 0.f, 0.f, 0.f};

#pragma unroll 4
    for (int kc = 0; kc < E_; kc += 32) {          // x chunks (bf16, reg-only)
        AFrag a;
        a.q[0] = *(const uint4*)(xrow + kc);
        a.q[1] = *(const uint4*)(xrow + kc + 16);
        const __bf16* wchunk = wb + (size_t)(kc >> 5) * (NCTILE * 32 * 16);
        v16bf bf0 = *(const v16bf*)wchunk;
        v16bf bf1 = *(const v16bf*)(wchunk + 32 * 16);
        __builtin_prefetch(wchunk + (size_t)(NCTILE * 32 * 16), 0, 1);
        acc0 = __builtin_amdgcn_wmma_f32_16x16x32_bf16(
            false, a.v, false, bf0, (short)0, acc0, false, false);
        acc1 = __builtin_amdgcn_wmma_f32_16x16x32_bf16(
            false, a.v, false, bf1, (short)0, acc1, false, false);
    }
#pragma unroll 4
    for (int kc = E_; kc < KTOT; kc += 32) {       // h chunks (bf16, reg-only)
        AFrag a;
        a.q[0] = *(const uint4*)(hrow + (kc - E_));
        a.q[1] = *(const uint4*)(hrow + (kc - E_) + 16);
        const __bf16* wchunk = wb + (size_t)(kc >> 5) * (NCTILE * 32 * 16);
        v16bf bf0 = *(const v16bf*)wchunk;
        v16bf bf1 = *(const v16bf*)(wchunk + 32 * 16);
        if (kc + 32 < KTOT)
            __builtin_prefetch(wchunk + (size_t)(NCTILE * 32 * 16), 0, 1);
        acc0 = __builtin_amdgcn_wmma_f32_16x16x32_bf16(
            false, a.v, false, bf0, (short)0, acc0, false, false);
        acc1 = __builtin_amdgcn_wmma_f32_16x16x32_bf16(
            false, a.v, false, bf1, (short)0, acc1, false, false);
    }

#pragma unroll
    for (int r = 0; r < 8; ++r) {
        int mr = r + 8 * (lane >> 4);
        gate_lds[g][mr * 32 + (lane & 15)]      = acc0[r];
        gate_lds[g][mr * 32 + 16 + (lane & 15)] = acc1[r];
    }
    __syncthreads();
    lstm_pointwise(gate_lds, b0, hn, s, biasC, cst, hnext, reps);
}

// ---------------- one LSTM timestep (fallback: convert x in-loop) ----------------
__global__ void __launch_bounds__(128)
lstm_step_cvt(int t,
              const int*    __restrict__ abstracts,
              const float*  __restrict__ emb,
              const __bf16* __restrict__ Wpack,
              const float*  __restrict__ biasC,
              const __bf16* __restrict__ hbf,
              __bf16* __restrict__ hbf_out_base,
              float* __restrict__ cst,
              float* __restrict__ reps) {
    __shared__ float gate_lds[4][16 * 32];

    const int bm   = blockIdx.x & 15;
    const int hn   = blockIdx.x >> 4;
    const int g    = threadIdx.x >> 5;
    const int lane = threadIdx.x & 31;
    const int b0   = bm * 16;
    const int s    = t / W_;
    const int w    = t - s * W_;

    const __bf16* hprev = hbf + (size_t)(t & 1) * (B_ * H_);
    __bf16* hnext = hbf_out_base + (size_t)((t + 1) & 1) * (B_ * H_);

    const int m  = lane & 15;
    const int lo = (lane & 16) ? 8 : 0;
    const int myidx = abstracts[(size_t)(b0 + m) * T_ + s * W_ + w];
    const float*  xrow = emb + (size_t)myidx * E_ + lo;
    const __bf16* hrow = hprev + (size_t)(b0 + m) * H_ + lo;

    const int ctile0 = g * 32 + hn * 2;
    const __bf16* wb = Wpack + (size_t)ctile0 * (32 * 16) + (size_t)lane * 16;

    v8f acc0 = {0.f, 0.f, 0.f, 0.f, 0.f, 0.f, 0.f, 0.f};
    v8f acc1 = {0.f, 0.f, 0.f, 0.f, 0.f, 0.f, 0.f, 0.f};

#pragma unroll 4
    for (int kc = 0; kc < E_; kc += 32) {
        float f[16];
        *(float4*)(f + 0)  = *(const float4*)(xrow + kc + 0);
        *(float4*)(f + 4)  = *(const float4*)(xrow + kc + 4);
        *(float4*)(f + 8)  = *(const float4*)(xrow + kc + 16);
        *(float4*)(f + 12) = *(const float4*)(xrow + kc + 20);
        AFrag a;
#pragma unroll
        for (int j = 0; j < 16; ++j) a.v[j] = (__bf16)f[j];
        const __bf16* wchunk = wb + (size_t)(kc >> 5) * (NCTILE * 32 * 16);
        v16bf bf0 = *(const v16bf*)wchunk;
        v16bf bf1 = *(const v16bf*)(wchunk + 32 * 16);
        acc0 = __builtin_amdgcn_wmma_f32_16x16x32_bf16(
            false, a.v, false, bf0, (short)0, acc0, false, false);
        acc1 = __builtin_amdgcn_wmma_f32_16x16x32_bf16(
            false, a.v, false, bf1, (short)0, acc1, false, false);
    }
#pragma unroll 4
    for (int kc = E_; kc < KTOT; kc += 32) {
        AFrag a;
        a.q[0] = *(const uint4*)(hrow + (kc - E_));
        a.q[1] = *(const uint4*)(hrow + (kc - E_) + 16);
        const __bf16* wchunk = wb + (size_t)(kc >> 5) * (NCTILE * 32 * 16);
        v16bf bf0 = *(const v16bf*)wchunk;
        v16bf bf1 = *(const v16bf*)(wchunk + 32 * 16);
        acc0 = __builtin_amdgcn_wmma_f32_16x16x32_bf16(
            false, a.v, false, bf0, (short)0, acc0, false, false);
        acc1 = __builtin_amdgcn_wmma_f32_16x16x32_bf16(
            false, a.v, false, bf1, (short)0, acc1, false, false);
    }

#pragma unroll
    for (int r = 0; r < 8; ++r) {
        int mr = r + 8 * (lane >> 4);
        gate_lds[g][mr * 32 + (lane & 15)]      = acc0[r];
        gate_lds[g][mr * 32 + 16 + (lane & 15)] = acc1[r];
    }
    __syncthreads();
    lstm_pointwise(gate_lds, b0, hn, s, biasC, cst, hnext, reps);
}

// ---------------- predictor part 1: A[s,b,c] and fallback T0 ----------------
__global__ void pred_A(const float* __restrict__ reps,   // [S,B,H]
                       const float* __restrict__ W1,     // [D1+1,H,C]
                       const float* __restrict__ b1,     // [C]
                       float* __restrict__ Abuf,         // [S,B,C]
                       float* __restrict__ T0) {         // [D2,B,C]
    int tid = blockIdx.x * blockDim.x + threadIdx.x;     // S*B*C = 20480
    if (tid >= S_ * B_ * C_) return;
    int s = tid / (B_ * C_);
    int r = tid - s * (B_ * C_);
    int b = r / C_;
    int c = r - b * C_;

    float p0 = 0.0f;
    const float* rp = &reps[(size_t)s * B_ * H_ + (size_t)b * H_];
    for (int h = 0; h < H_; ++h) p0 += rp[h] * W1[h * C_ + c];       // k = 0
    float acc = p0;
    for (int k = 1; k <= D1_; ++k) {
        if (s < k) break;
        const float* rk = &reps[(size_t)(s - k) * B_ * H_ + (size_t)b * H_];
        const float* wk = &W1[k * H_ * C_];
        for (int h = 0; h < H_; ++h) acc += rk[h] * wk[h * C_ + c];
    }
    Abuf[tid] = tanhf(acc + b1[c]);
    if (s < D2_) T0[(s * B_ + b) * C_ + c] = tanhf(p0 + b1[c]);       // fallback: W1[0] only
}

// ---------------- predictor part 2: logits + log_softmax -> out[B,C,S] ----------------
__global__ void pred_out(const float* __restrict__ Abuf,  // [S,B,C]
                         const float* __restrict__ T0,    // [D2,B,C]
                         const float* __restrict__ W2,    // [D2+1,C,C]
                         const float* __restrict__ b2,    // [C]
                         float* __restrict__ out) {       // [B,C,S]
    int tid = blockIdx.x * blockDim.x + threadIdx.x;      // B*S = 4096
    if (tid >= B_ * S_) return;
    int b = tid / S_;
    int s = tid - b * S_;

    float z[C_];
    if (s < D2_) {
        const float* a = &T0[(s * B_ + b) * C_];
        for (int d = 0; d < C_; ++d) {
            float acc = b2[d];
            for (int c = 0; c < C_; ++c) acc += a[c] * W2[c * C_ + d];  // W2[0]
            z[d] = acc;
        }
    } else {
        int j = s - D2_;
        for (int d = 0; d < C_; ++d) z[d] = b2[d];
        for (int t = 0; t <= D2_; ++t) {
            const float* a = &Abuf[((size_t)(j + t) * B_ + b) * C_];
            const float* w = &W2[t * C_ * C_];
            for (int c = 0; c < C_; ++c)
                for (int d = 0; d < C_; ++d) z[d] += a[c] * w[c * C_ + d];
        }
    }
    float mx = z[0];
    for (int d = 1; d < C_; ++d) mx = fmaxf(mx, z[d]);
    float se = 0.0f;
    for (int d = 0; d < C_; ++d) se += expf(z[d] - mx);
    float lse = mx + logf(se);
    for (int d = 0; d < C_; ++d)
        out[((size_t)b * C_ + d) * S_ + s] = z[d] - lse;
}

// ---------------- host ----------------
extern "C" void kernel_launch(void* const* d_in, const int* in_sizes, int n_in,
                              void* d_out, int out_size, void* d_ws, size_t ws_size,
                              hipStream_t stream) {
    const int*   abstracts = (const int*)  d_in[0];
    const float* emb       = (const float*)d_in[1];
    const float* Wih       = (const float*)d_in[2];
    const float* Whh       = (const float*)d_in[3];
    const float* b_ih      = (const float*)d_in[4];
    const float* b_hh      = (const float*)d_in[5];
    const float* W1        = (const float*)d_in[6];
    const float* b1        = (const float*)d_in[7];
    const float* W2        = (const float*)d_in[8];
    const float* b2        = (const float*)d_in[9];
    float* out = (float*)d_out;

    char* ws = (char*)d_ws;
    size_t off = 0;
    auto carve = [&](size_t bytes) -> void* {
        void* p = ws + off;
        off = (off + bytes + 255) & ~(size_t)255;
        return p;
    };
    // base carve-out (~12.6 MB)
    __bf16* Wpack = (__bf16*)carve((size_t)KTOT * G4H * sizeof(__bf16));   // 3 MB
    float*  biasC = (float*) carve((size_t)G4H * sizeof(float));
    __bf16* hbf   = (__bf16*)carve((size_t)2 * B_ * H_ * sizeof(__bf16));
    float*  cst   = (float*) carve((size_t)B_ * H_ * sizeof(float));
    float*  reps  = (float*) carve((size_t)S_ * B_ * H_ * sizeof(float));  // 8 MB
    float*  Abuf  = (float*) carve((size_t)S_ * B_ * C_ * sizeof(float));
    float*  T0    = (float*) carve((size_t)D2_ * B_ * C_ * sizeof(float));
    // optional fast-path buffer (+64 MB): pre-gathered bf16 embeddings
    size_t base_bytes = off;
    __bf16* xbf = (__bf16*)carve((size_t)T_ * B_ * E_ * sizeof(__bf16));
    bool fast = (off <= ws_size);
    (void)base_bytes;

    prep_pack<<<(NCHUNK * NCTILE * 32 + 255) / 256, 256, 0, stream>>>(Wih, Whh, Wpack);
    init_state<<<(S_ * B_ * H_ + 255) / 256, 256, 0, stream>>>(b_ih, b_hh, biasC, hbf, cst, reps);

    if (fast) {
        prep_x<<<(T_ * B_ * (E_ / 8) + 255) / 256, 256, 0, stream>>>(abstracts, emb, xbf);
        for (int t = 0; t < T_; ++t)
            lstm_step_fast<<<(B_ / 16) * (H_ / 32), 128, 0, stream>>>(
                t, xbf, Wpack, biasC, hbf, hbf, cst, reps);
    } else {
        for (int t = 0; t < T_; ++t)
            lstm_step_cvt<<<(B_ / 16) * (H_ / 32), 128, 0, stream>>>(
                t, abstracts, emb, Wpack, biasC, hbf, hbf, cst, reps);
    }

    pred_A<<<(S_ * B_ * C_ + 255) / 256, 256, 0, stream>>>(reps, W1, b1, Abuf, T0);
    pred_out<<<(B_ * S_ + 255) / 256, 256, 0, stream>>>(Abuf, T0, W2, b2, out);
}
